// ChannelAttentionBlock_55224689492543
// MI455X (gfx1250) — compile-verified, matrix operand order checked by reading, not run on hardware
//
#include <hip/hip_runtime.h>
#include <hip/hip_bf16.h>

// ---------------------------------------------------------------------------
// Problem constants (from reference): x [B=32, C=512, H=56, W=56] fp32
// ---------------------------------------------------------------------------
#define BATCH 32
#define CH    512
#define HW    56
#define PLANE (HW * HW)      // 3136
#define KP    64             // K padded 56 -> 64 (zeros)
#define FEAT_ELEMS (CH * KP) // 32768 floats per (type,batch)

typedef __attribute__((ext_vector_type(16))) __bf16 v16bf;
typedef __attribute__((ext_vector_type(8)))  __bf16 v8bf;
typedef __attribute__((ext_vector_type(8)))  float  v8f;

// ---------------------------------------------------------------------------
// Kernel 1: adaptive max pools.  One block per (b,c) plane.
//   type 0: xh[l] = max_w x[.., l, w]   (pool width)
//   type 1: xw[l] = max_h x[.., h, l]   (pool height)
// Output zero-padded to KP=64 so padded K never perturbs dot products.
// ---------------------------------------------------------------------------
__global__ void __launch_bounds__(256) pool_kernel(const float* __restrict__ x,
                                                   float* __restrict__ featp) {
    const int plane = blockIdx.x;              // b*512 + c
    const int t = threadIdx.x;
    __shared__ float pl[PLANE];

    const float* xp = x + (size_t)plane * PLANE;
    for (int i = t; i < PLANE; i += 256) pl[i] = xp[i];
    __syncthreads();

    if (t < 64) {                               // row max -> xh (type 0)
        const int l = t;
        float m = 0.0f;
        if (l < HW) {
            m = -3.402823466e38f;
            #pragma unroll 8
            for (int w = 0; w < HW; ++w) m = fmaxf(m, pl[l * HW + w]);
        }
        featp[(size_t)plane * KP + l] = m;
    } else if (t < 128) {                       // col max -> xw (type 1)
        const int l = t - 64;
        float m = 0.0f;
        if (l < HW) {
            m = -3.402823466e38f;
            #pragma unroll 8
            for (int h = 0; h < HW; ++h) m = fmaxf(m, pl[h * HW + l]);
        }
        featp[(size_t)(BATCH * CH + plane) * KP + l] = m;
    }
}

// ---------------------------------------------------------------------------
// WMMA fragment loader: gather the two non-contiguous 16B runs of a
// 16-bit 16x32 operand (V0-3 = K base+half*8, V4-7 = K base+16+half*8).
// ---------------------------------------------------------------------------
static __device__ __forceinline__ v16bf load_frag(const __bf16* lo, const __bf16* hi) {
    v8bf a = *(const v8bf*)lo;
    v8bf b = *(const v8bf*)hi;
    v16bf r;
    #pragma unroll
    for (int i = 0; i < 8; ++i) { r[i] = a[i]; r[i + 8] = b[i]; }
    return r;
}

// ---------------------------------------------------------------------------
// Kernel 2: channel attention via v_wmma_f32_16x16x32_bf16.
//   grid = (64, 4): blockIdx.x = type*32+b, blockIdx.y = row-tile group.
//   8 waves / block; wave w owns row tile ct = blockIdx.y*8 + w (16 rows).
//   softmax(rowmax - aff) == exp(rowmin - aff) / sum  (shift cancels).
// ---------------------------------------------------------------------------
__global__ void __launch_bounds__(256) attn_kernel(const float* __restrict__ featp,
                                                   float* __restrict__ attn) {
    __shared__ __bf16 featA[CH * KP];    // 64 KB: feat rows (A and aff-B operands)
    __shared__ __bf16 featT[KP * CH];    // 64 KB: feat transposed (numerator B)
    __shared__ __bf16 pbuf[8 * 16 * 32]; // 8 KB : per-wave P re-fragment bounce

    const int tb   = blockIdx.x;                 // type*32 + b
    const int tid  = threadIdx.x;
    const int wave = tid >> 5;
    const int lane = tid & 31;
    const int half = lane >> 4;                  // K split within fragment
    const int lm   = lane & 15;                  // row/col within fragment

    const float* fsrc = featp + (size_t)tb * FEAT_ELEMS;
    for (int i = tid; i < FEAT_ELEMS; i += 256) {
        const float v = fsrc[i];
        const __bf16 h = (__bf16)v;
        const int c = i >> 6, l = i & 63;
        featA[i] = h;
        featT[l * CH + c] = h;
    }
    __syncthreads();

    const int ct = blockIdx.y * 8 + wave;        // row tile 0..31
    const int R0 = ct * 16;

    // A fragments for this wave's 16 rows (K = 0..63), constant across loops.
    const __bf16* arow = featA + (R0 + lm) * KP;
    const v16bf a0 = load_frag(arow + half * 8,      arow + 16 + half * 8);
    const v16bf a1 = load_frag(arow + 32 + half * 8, arow + 48 + half * 8);

    // ---- Phase 1: row minimum of aff over all 512 columns -----------------
    float rmin[8];
    #pragma unroll
    for (int j = 0; j < 8; ++j) rmin[j] = 3.402823466e38f;

    for (int dt = 0; dt < 32; ++dt) {
        const __bf16* brow = featA + (dt * 16 + lm) * KP;
        const v16bf b0 = load_frag(brow + half * 8,      brow + 16 + half * 8);
        const v16bf b1 = load_frag(brow + 32 + half * 8, brow + 48 + half * 8);
        v8f acc = {};
        acc = __builtin_amdgcn_wmma_f32_16x16x32_bf16(false, a0, false, b0,
                                                      (short)0, acc, false, false);
        acc = __builtin_amdgcn_wmma_f32_16x16x32_bf16(false, a1, false, b1,
                                                      (short)0, acc, false, false);
        #pragma unroll
        for (int j = 0; j < 8; ++j) rmin[j] = fminf(rmin[j], acc[j]);
    }
    #pragma unroll
    for (int j = 0; j < 8; ++j) {
        #pragma unroll
        for (int s = 1; s < 16; s <<= 1)         // stays within 16-lane halves
            rmin[j] = fminf(rmin[j], __shfl_xor(rmin[j], s, 32));
    }

    // ---- Phase 2: P = exp(rmin - aff); numerator = P @ feat ---------------
    float rsum[8];
    #pragma unroll
    for (int j = 0; j < 8; ++j) rsum[j] = 0.0f;
    v8f accn[4];
    #pragma unroll
    for (int lt = 0; lt < 4; ++lt) accn[lt] = (v8f){};

    __bf16* pb = pbuf + wave * (16 * 32);

    for (int dt2 = 0; dt2 < 16; ++dt2) {
        #pragma unroll
        for (int sub = 0; sub < 2; ++sub) {
            const int dt = dt2 * 2 + sub;
            const __bf16* brow = featA + (dt * 16 + lm) * KP;
            const v16bf b0 = load_frag(brow + half * 8,      brow + 16 + half * 8);
            const v16bf b1 = load_frag(brow + 32 + half * 8, brow + 48 + half * 8);
            v8f acc = {};
            acc = __builtin_amdgcn_wmma_f32_16x16x32_bf16(false, a0, false, b0,
                                                          (short)0, acc, false, false);
            acc = __builtin_amdgcn_wmma_f32_16x16x32_bf16(false, a1, false, b1,
                                                          (short)0, acc, false, false);
            #pragma unroll
            for (int j = 0; j < 8; ++j) {
                const float p = __expf(rmin[j] - acc[j]);
                rsum[j] += p;
                // C layout: VGPR j, lanes 0-15 -> row j, lanes 16-31 -> row j+8
                pb[(j + half * 8) * 32 + sub * 16 + lm] = (__bf16)p;
            }
        }
        // Re-fragment P (16 rows x 32 Ks) as an A operand from LDS.
        const __bf16* prow = pb + lm * 32;
        const v16bf ap = load_frag(prow + half * 8, prow + 16 + half * 8);
        const int dbase = dt2 * 32;
        #pragma unroll
        for (int lt = 0; lt < 4; ++lt) {
            const __bf16* trow = featT + (lt * 16 + lm) * CH + dbase;
            const v16bf bp = load_frag(trow + half * 8, trow + 16 + half * 8);
            accn[lt] = __builtin_amdgcn_wmma_f32_16x16x32_bf16(false, ap, false, bp,
                                                               (short)0, accn[lt],
                                                               false, false);
        }
    }

    #pragma unroll
    for (int j = 0; j < 8; ++j) {
        #pragma unroll
        for (int s = 1; s < 16; s <<= 1)
            rsum[j] += __shfl_xor(rsum[j], s, 32);
    }

    float* adst = attn + (size_t)tb * FEAT_ELEMS;
    #pragma unroll
    for (int j = 0; j < 8; ++j) {
        const float inv = 1.0f / rsum[j];
        const int row = R0 + j + half * 8;
        #pragma unroll
        for (int lt = 0; lt < 4; ++lt)
            adst[row * KP + lt * 16 + lm] = accn[lt][j] * inv;
    }
}

// ---------------------------------------------------------------------------
// Kernel 3: out = gamma * w1[h] * x * w2[w] + x, float4 streaming.
// One block per (b,c) plane: 784 float4 per plane, 16B aligned.
// ---------------------------------------------------------------------------
__global__ void __launch_bounds__(256) combine_kernel(const float* __restrict__ x,
                                                      const float* __restrict__ gamma,
                                                      const float* __restrict__ attn,
                                                      float* __restrict__ out) {
    const int plane = blockIdx.x;                // b*512 + c
    const int t = threadIdx.x;
    const float g = gamma[0];

    const float* w1p = attn + (size_t)plane * KP;                      // type 0 (h)
    const float* w2p = attn + (size_t)(BATCH * CH + plane) * KP;       // type 1 (w)
    const float4* xp4 = (const float4*)(x + (size_t)plane * PLANE);
    float4* op4       = (float4*)(out + (size_t)plane * PLANE);

    #pragma unroll
    for (int k = 0; k < 4; ++k) {
        const int i = t + k * 256;               // float4 index within plane
        if (i < PLANE / 4) {
            const int h  = i / (HW / 4);         // 14 float4 per row
            const int wb = (i % (HW / 4)) * 4;
            const float4 xv = xp4[i];
            const float  gw1 = g * w1p[h];
            const float4 w2  = *(const float4*)(w2p + wb);
            float4 o;
            o.x = fmaf(gw1 * w2.x, xv.x, xv.x);
            o.y = fmaf(gw1 * w2.y, xv.y, xv.y);
            o.z = fmaf(gw1 * w2.z, xv.z, xv.z);
            o.w = fmaf(gw1 * w2.w, xv.w, xv.w);
            op4[i] = o;
        }
    }
}

// ---------------------------------------------------------------------------
extern "C" void kernel_launch(void* const* d_in, const int* in_sizes, int n_in,
                              void* d_out, int out_size, void* d_ws, size_t ws_size,
                              hipStream_t stream) {
    const float* x     = (const float*)d_in[0];
    const float* gamma = (const float*)d_in[1];
    float* out = (float*)d_out;

    float* featp = (float*)d_ws;                       // [2][32][512][64] = 8 MB
    float* attn  = featp + 2 * BATCH * CH * KP;        // [2][32][512][64] = 8 MB

    pool_kernel<<<BATCH * CH, 256, 0, stream>>>(x, featp);

    dim3 g2(2 * BATCH, 4);
    attn_kernel<<<g2, 256, 0, stream>>>(featp, attn);

    combine_kernel<<<BATCH * CH, 256, 0, stream>>>(x, gamma, attn, out);
}
